// DynamicWeights__17721035064090
// MI455X (gfx1250) — matrix operand order repaired and checked by reading, not compiled, not run on hardware
//
#include <hip/hip_runtime.h>
#include <hip/hip_bf16.h>
#include <math.h>

// ---------------------------------------------------------------------------
// DynamicWeights on MI455X (gfx1250, wave32, WMMA)
//
//   offs = conv5x5(cat(query,support), w_off)   ; GEMM M=18 K=12800 Npix=18432
//   val  = bilinear-sample(support, offs)       ; gather
//   out  = sigmoid(einsum(val, w_kernel))       ; GEMM M=9  K=2304  Npix=18432
//
// Both GEMMs on v_wmma_f32_16x16x32_bf16 (f32 accumulate). Weights are
// pre-packed into a 16-bit A-fragment layout with K=(L>>4)*16+e (32B per lane per chunk).
// conv_offset: double-buffered LDS B tile, software-pipelined, 1 barrier per
// k-chunk, BRANCHLESS zero-padded staging (clamped address + mask) so the 8
// loads per thread issue as one clause. deform_gemm: B fragment built
// directly in registers (lane L owns pixel L&15 / K-half L>>4 of the 32x16
// B layout) -> no LDS, no barriers.
// ---------------------------------------------------------------------------

typedef __attribute__((ext_vector_type(16))) __bf16 v16bf;
typedef __attribute__((ext_vector_type(8)))  __bf16 v8bf;
typedef __attribute__((ext_vector_type(8)))  float  v8f;

#define Nn   8
#define Cc   256
#define Hh   48
#define Ww   48
#define HW   (Hh * Ww)
#define KT   9          // 3x3 deform taps
#define OCO  18         // offset channels (2*KT)
#define KC1  400        // k-chunks offset conv: 25 taps * (512/32)
#define KC2  72         // k-chunks deform gemm:  9 taps * (256/32)

static __device__ __forceinline__ int clampi(int v, int lo, int hi) {
  return v < lo ? lo : (v > hi ? hi : v);
}

// A-fragment element->K map, 16-bit 16x32 A tile (matches B K order):
// lane L: M = L&15, kbase = (L>>4)*16 ; elem e: K = kbase+e
static __device__ __forceinline__ int a_elem_k(int lane, int e) {
  int kbase = (lane >> 4) * 16;
  return kbase + e;
}

// --------------------------- weight pack kernels ---------------------------

// w_off [18,512,5,5] -> apack1[mtile=2][chunk=400][lane=32]; k = tap*512 + c
__global__ void pack_woff(const float* __restrict__ w_off,
                          v16bf* __restrict__ apack) {
  const int chunk = blockIdx.x;
  const int mtile = blockIdx.y;
  const int lane  = threadIdx.x;
  const int m   = (lane & 15) + mtile * 16;
  const int tap = chunk >> 4;
  const int cb  = (chunk & 15) * 32;
  const int dy5 = tap / 5, dx5 = tap % 5;
  v16bf v;
#pragma unroll
  for (int e = 0; e < 16; ++e) {
    int c = cb + a_elem_k(lane, e);
    float f = 0.0f;
    if (m < OCO)
      f = w_off[((m * 512 + c) * 5 + dy5) * 5 + dx5];
    v[e] = (__bf16)f;
  }
  apack[(mtile * KC1 + chunk) * 32 + lane] = v;
}

// w_kernel [9,256,3,3] -> apack2[chunk=72][lane=32]; k = tap*256 + c
__global__ void pack_wker(const float* __restrict__ w_ker,
                          v16bf* __restrict__ apack) {
  const int chunk = blockIdx.x;
  const int lane  = threadIdx.x;
  const int m   = lane & 15;
  const int tap = chunk >> 3;
  const int cb  = (chunk & 7) * 32;
  const int ky  = tap / 3, kx = tap % 3;
  v16bf v;
#pragma unroll
  for (int e = 0; e < 16; ++e) {
    int c = cb + a_elem_k(lane, e);
    float f = 0.0f;
    if (m < KT)
      f = w_ker[((m * 256 + c) * 3 + ky) * 3 + kx];
    v[e] = (__bf16)f;
  }
  apack[chunk * 32 + lane] = v;
}

// --------------------------- offset 5x5 conv (WMMA) ------------------------
// Block = 16 pixels (n,h,wtile), 2 waves (wave = M-tile). Double-buffered
// LDS B tile, software pipelined: stage chunk+1 while WMMA eats chunk.
__global__ void conv_offset(const float* __restrict__ query,
                            const float* __restrict__ support,
                            const v16bf* __restrict__ apack,
                            float* __restrict__ offs) {
  __shared__ __align__(128) __bf16 Bs[2][16][32];   // [buf][pixel][klocal]

  const int wtile = blockIdx.x;            // 0..2
  const int h     = blockIdx.y;            // 0..47
  const int n     = blockIdx.z;            // 0..7
  const int t     = threadIdx.x;           // 0..63
  const int lane  = t & 31;
  const int wv    = t >> 5;                // M-tile id
  const int wbase = wtile * 16;
  const int p     = t & 15;                // staging pixel
  const int kg    = t >> 4;                // 0..3 staging k-group

  // tap (dy,dx) of the chunk currently being STAGED; carried incrementally
  int sdy = -2, sdx = -2;

  auto stage = [&](int ch, int buf) {
    const int  cb  = (ch & 15) * 32;
    const int  iy  = h + sdy;
    const int  ix  = wbase + p + sdx;
    // branchless zero padding: clamp address into the image, mask the value
    const float msk =
        ((iy >= 0) & (iy < Hh) & (ix >= 0) & (ix < Ww)) ? 1.0f : 0.0f;
    const int  iyc = clampi(iy, 0, Hh - 1);
    const int  ixc = clampi(ix, 0, Ww - 1);
    const float* src = (cb < Cc) ? query : support;        // cat([q, s])
    const int  ccb  = (cb & (Cc - 1)) + kg * 8;
    const float* gp = src + ((size_t)(n * Cc + ccb) * Hh + iyc) * Ww + ixc;
    v8bf pk;
#pragma unroll
    for (int i = 0; i < 8; ++i)            // 8 unconditional clause'd loads
      pk[i] = (__bf16)(gp[(size_t)i * HW] * msk);
    *reinterpret_cast<v8bf*>(&Bs[buf][p][kg * 8]) = pk;    // one b128 store
  };

  stage(0, 0);
  __syncthreads();

  v8f acc = {};
  for (int chunk = 0; chunk < KC1; ++chunk) {
    const int nxt = chunk + 1;
    if (nxt < KC1) {
      if ((nxt & 15) == 0) { if (++sdx == 3) { sdx = -2; ++sdy; } }
      stage(nxt, nxt & 1);
      __builtin_prefetch(&apack[(wv * KC1 + nxt) * 32 + lane], 0, 1);
    }
    const v16bf a = apack[(wv * KC1 + chunk) * 32 + lane];
    const v16bf* row =
        reinterpret_cast<const v16bf*>(&Bs[chunk & 1][lane & 15][0]);
    const v16bf b = row[lane >> 4];        // 32B contiguous: K=(L>>4)*16+e
    acc = __builtin_amdgcn_wmma_f32_16x16x32_bf16(
        false, a, false, b, (short)0, acc, false, false);
    __syncthreads();                       // publish next buf / retire reads
  }

  // C layout: VGPR r, lanes0-15: M=r ; lanes16-31: M=8+r ; N=lane&15
  const int colw = wbase + (lane & 15);
  const int mb   = wv * 16 + 8 * (lane >> 4);
#pragma unroll
  for (int r = 0; r < 8; ++r) {
    const int M = mb + r;
    if (M < OCO)
      offs[((n * OCO + M) * Hh + h) * Ww + colw] = acc[r];
  }
}

// ------------------- deformable sampling + einsum (WMMA) -------------------
// 256-thread blocks, 8 fully independent waves, one 16-pixel tile per wave.
// Lane L computes bilinear values for pixel L&15, channels (L>>4)*16+e —
// which IS its 32x16 B fragment, so it goes straight into registers.
__global__ void deform_gemm(const float* __restrict__ support,
                            const float* __restrict__ offs,
                            const v16bf* __restrict__ apack,
                            float* __restrict__ out) {
  const int t    = threadIdx.x;            // 0..255
  const int wv   = t >> 5;                 // 0..7
  const int lane = t & 31;
  const int tile = blockIdx.x * 8 + wv;    // 0..1151
  const int n     = tile / (Hh * 3);
  const int rem   = tile % (Hh * 3);
  const int h     = rem / 3;
  const int wbase = (rem % 3) * 16;

  const int p     = lane & 15;             // this lane's pixel
  const int w     = wbase + p;
  const int chalf = lane >> 4;             // K-half within chunk
  const float* simg = support + (size_t)n * Cc * HW;

  v8f acc = {};
  int chunk = 0;
#pragma unroll
  for (int ky = 0; ky < 3; ++ky) {
#pragma unroll
    for (int kx = 0; kx < 3; ++kx) {
      const int tap = ky * 3 + kx;
      // per-tap bilinear params, computed per lane (registers only)
      const float dy = offs[((n * OCO + 2 * tap)     * Hh + h) * Ww + w];
      const float dx = offs[((n * OCO + 2 * tap + 1) * Hh + h) * Ww + w];
      const float sy = (float)(h + ky - 1) + dy;       // pad = dil = 1
      const float sx = (float)(w + kx - 1) + dx;
      const float y0f = floorf(sy), x0f = floorf(sx);
      const float ly = sy - y0f, lx = sx - x0f;
      const int y0 = (int)y0f, x0 = (int)x0f;
      const int y1 = y0 + 1,  x1 = x0 + 1;
      const float vy0 = (y0 >= 0 && y0 < Hh) ? 1.f : 0.f;
      const float vy1 = (y1 >= 0 && y1 < Hh) ? 1.f : 0.f;
      const float vx0 = (x0 >= 0 && x0 < Ww) ? 1.f : 0.f;
      const float vx1 = (x1 >= 0 && x1 < Ww) ? 1.f : 0.f;
      const float w00 = (1.f - ly) * (1.f - lx) * vy0 * vx0;
      const float w01 = (1.f - ly) * lx         * vy0 * vx1;
      const float w10 = ly         * (1.f - lx) * vy1 * vx0;
      const float w11 = ly         * lx         * vy1 * vx1;
      const int y0c = clampi(y0, 0, Hh - 1), x0c = clampi(x0, 0, Ww - 1);
      const int y1c = clampi(y1, 0, Hh - 1), x1c = clampi(x1, 0, Ww - 1);
      const int a00 = y0c * Ww + x0c, a01 = y0c * Ww + x1c;
      const int a10 = y1c * Ww + x0c, a11 = y1c * Ww + x1c;

      for (int cc = 0; cc < 8; ++cc, ++chunk) {
        // B fragment directly in registers: c = cc*32 + chalf*16 + e
        const float* base = simg + (size_t)(cc * 32 + chalf * 16) * HW;
        v16bf b;
#pragma unroll
        for (int i = 0; i < 16; ++i) {
          const float* bp = base + (size_t)i * HW;
          const float v = w00 * bp[a00] + w01 * bp[a01] +
                          w10 * bp[a10] + w11 * bp[a11];
          b[i] = (__bf16)v;
        }
        if (chunk + 1 < KC2)
          __builtin_prefetch(&apack[(chunk + 1) * 32 + lane], 0, 1);
        const v16bf a = apack[chunk * 32 + lane];
        acc = __builtin_amdgcn_wmma_f32_16x16x32_bf16(
            false, a, false, b, (short)0, acc, false, false);
      }
    }
  }

  // fused sigmoid epilogue; only M<9 valid
  const int colw = wbase + (lane & 15);
  const int mb   = 8 * (lane >> 4);
#pragma unroll
  for (int r = 0; r < 8; ++r) {
    const int o = mb + r;
    if (o < KT)
      out[((n * KT + o) * Hh + h) * Ww + colw] =
          1.0f / (1.0f + __expf(-acc[r]));
  }
}

// ------------------------------- launcher ----------------------------------

extern "C" void kernel_launch(void* const* d_in, const int* in_sizes, int n_in,
                              void* d_out, int out_size, void* d_ws, size_t ws_size,
                              hipStream_t stream) {
  const float* support = (const float*)d_in[0];  // [8,256,48,48]
  const float* query   = (const float*)d_in[1];  // [8,256,48,48]
  const float* w_off   = (const float*)d_in[2];  // [18,512,5,5]
  const float* w_ker   = (const float*)d_in[3];  // [9,256,3,3]
  float* out = (float*)d_out;                    // [8,9,48,48]

  // workspace layout (32B-aligned offsets)
  char* ws = (char*)d_ws;
  float* offs   = (float*)ws;                            // 1,327,104 B
  v16bf* apack1 = (v16bf*)(ws + 1327104);                //   819,200 B
  v16bf* apack2 = (v16bf*)(ws + 1327104 + 819200);       //    73,728 B

  pack_woff  <<<dim3(KC1, 2),    32, 0, stream>>>(w_off, apack1);
  pack_wker  <<<dim3(KC2),       32, 0, stream>>>(w_ker, apack2);
  conv_offset<<<dim3(3, Hh, Nn), 64, 0, stream>>>(query, support, apack1, offs);
  deform_gemm<<<dim3(144),      256, 0, stream>>>(support, offs, apack2, out);
}
